// SoftAttention_1262720385456
// MI455X (gfx1250) — compile-verified, matrix operand order checked by reading, not run on hardware
//
#include <hip/hip_runtime.h>
#include <hip/hip_bf16.h>

// Problem sizes (fixed by the reference)
#define BATCH   64
#define FEATN   512
#define FEATSZ  2048
#define HID     1024
#define ATT     512

typedef __attribute__((ext_vector_type(16))) __bf16 v16bf;
typedef __attribute__((ext_vector_type(8)))  float  v8f;

union V16BF {
    v16bf v;
    uint4 q[2];            // two 16-byte halves (8 bf16 each)
    unsigned short us[16];
};

// ---- fast tanh: native v_tanh_f32 on gfx1250, branch-free fallback ----
#if __has_builtin(__builtin_amdgcn_tanhf)
__device__ __forceinline__ float fast_tanh(float x) { return __builtin_amdgcn_tanhf(x); }
#elif __has_builtin(__builtin_amdgcn_tanh_f32)
__device__ __forceinline__ float fast_tanh(float x) { return __builtin_amdgcn_tanh_f32(x); }
#else
__device__ __forceinline__ float fast_tanh(float x) {
    // tanh(x) = sign(x) * (1 - e) / (1 + e), e = exp2(-2|x| * log2(e))
    float ax = __builtin_fabsf(x);
    float e  = __builtin_amdgcn_exp2f(ax * -2.8853900817779268f);
    float r  = (1.0f - e) * __builtin_amdgcn_rcpf(1.0f + e);
    return __builtin_copysignf(r, x);
}
#endif

__device__ __forceinline__ float fast_exp(float x) {
    return __builtin_amdgcn_exp2f(x * 1.4426950408889634f);
}

// ---------------------------------------------------------------------------
// Kernel 1: convert wv_w (ATT x FEATSZ fp32) -> bf16 in workspace
// ---------------------------------------------------------------------------
__global__ __launch_bounds__(256)
void cvt_bf16_kernel(const float* __restrict__ src, __bf16* __restrict__ dst) {
    int i = blockIdx.x * blockDim.x + threadIdx.x;     // one float4 per thread
    float4 v = ((const float4*)src)[i];
    union { __bf16 b[4]; uint2 q; } o;
    o.b[0] = (__bf16)v.x; o.b[1] = (__bf16)v.y;
    o.b[2] = (__bf16)v.z; o.b[3] = (__bf16)v.w;
    ((uint2*)dst)[i] = o.q;
}

// ---------------------------------------------------------------------------
// Kernel 2: hb[b][a] = key[b,:] . wh_w[a,:] + wh_b[a] + wv_b[a]
// ---------------------------------------------------------------------------
__global__ __launch_bounds__(256)
void hb_kernel(const float* __restrict__ key, const float* __restrict__ wh_w,
               const float* __restrict__ wh_b, const float* __restrict__ wv_b,
               float* __restrict__ hb) {
    int idx = blockIdx.x * blockDim.x + threadIdx.x;   // 0 .. BATCH*ATT-1
    int b = idx >> 9;
    int a = idx & (ATT - 1);
    const float* krow = key  + (size_t)b * HID;
    const float* wrow = wh_w + (size_t)a * HID;
    float s = 0.0f;
    #pragma unroll 4
    for (int j = 0; j < HID; j += 4) {
        float4 k4 = *(const float4*)(krow + j);
        float4 w4 = *(const float4*)(wrow + j);
        s = fmaf(k4.x, w4.x, s);
        s = fmaf(k4.y, w4.y, s);
        s = fmaf(k4.z, w4.z, s);
        s = fmaf(k4.w, w4.w, s);
    }
    hb[idx] = s + wh_b[a] + wv_b[a];
}

// ---------------------------------------------------------------------------
// Kernel 3 (WMMA): scores[b][n] = sum_a tanh(feats[b,n,:].wv_w[a,:] + hb[b,a]) * wa_w[a]
// Grid: (N/16, BATCH), block 256 = 8 waves. Wave w owns a-tiles [4w .. 4w+3].
// ---------------------------------------------------------------------------
__global__ __launch_bounds__(256)
void scores_kernel(const float* __restrict__ feats,
                   const __bf16* __restrict__ wvw_bf,
                   const float* __restrict__ hb,
                   const float* __restrict__ wa_w,
                   float* __restrict__ scores) {
    __shared__ float ssum[16];

    const int b    = blockIdx.y;
    const int n0   = blockIdx.x * 16;
    const int lane = threadIdx.x & 31;
    const int wave = threadIdx.x >> 5;
    const int m    = lane & 15;   // row within tile (A-matrix) / column (B, C)
    const int kh   = lane >> 4;   // K-half selector per ISA 16-bit layout

    if (threadIdx.x < 16) ssum[threadIdx.x] = 0.0f;
    __syncthreads();

    const float* arow = feats + ((size_t)b * FEATN + (n0 + m)) * FEATSZ;
    const int abase = wave * 64;

    v8f acc[4] = {};

    for (int kb = 0; kb < FEATSZ; kb += 32) {
        // ---- A fragment: 16x32 bf16 tile of feats, hardware-converted ----
        // lane holds K = kb + 8*kh + [0..7] in halves 0..7,
        //            K = kb + 16 + 8*kh + [0..7] in halves 8..15
        const float* p0 = arow + kb + kh * 8;
        const float* p1 = arow + kb + 16 + kh * 8;
        float4 a0 = *(const float4*)(p0);
        float4 a1 = *(const float4*)(p0 + 4);
        float4 a2 = *(const float4*)(p1);
        float4 a3 = *(const float4*)(p1 + 4);
        V16BF af;
        af.v[0]  = (__bf16)a0.x; af.v[1]  = (__bf16)a0.y;
        af.v[2]  = (__bf16)a0.z; af.v[3]  = (__bf16)a0.w;
        af.v[4]  = (__bf16)a1.x; af.v[5]  = (__bf16)a1.y;
        af.v[6]  = (__bf16)a1.z; af.v[7]  = (__bf16)a1.w;
        af.v[8]  = (__bf16)a2.x; af.v[9]  = (__bf16)a2.y;
        af.v[10] = (__bf16)a2.z; af.v[11] = (__bf16)a2.w;
        af.v[12] = (__bf16)a3.x; af.v[13] = (__bf16)a3.y;
        af.v[14] = (__bf16)a3.z; af.v[15] = (__bf16)a3.w;

        // ---- 4 B fragments (pre-converted bf16 wv_w), 4 WMMAs ----
        #pragma unroll
        for (int t = 0; t < 4; ++t) {
            const __bf16* brow = wvw_bf + (size_t)(abase + t * 16 + m) * FEATSZ;
            V16BF bf_;
            bf_.q[0] = *(const uint4*)(brow + kb + kh * 8);       // K halves 0..7
            bf_.q[1] = *(const uint4*)(brow + kb + 16 + kh * 8);  // K halves 8..15
            acc[t] = __builtin_amdgcn_wmma_f32_16x16x32_bf16(
                false, af.v, false, bf_.v, (short)0, acc[t], false, false);
        }
    }

    // ---- fused epilogue: tanh(v + hb) * wa_w, reduce over a ----
    // acc[t] element r: output row n = n0 + r + 8*kh, column a = abase + 16t + m
    float rowsum[8];
    #pragma unroll
    for (int r = 0; r < 8; ++r) rowsum[r] = 0.0f;

    #pragma unroll
    for (int t = 0; t < 4; ++t) {
        const int a = abase + t * 16 + m;
        const float bias = hb[(size_t)b * ATT + a];
        const float wt   = wa_w[a];
        #pragma unroll
        for (int r = 0; r < 8; ++r)
            rowsum[r] += fast_tanh(acc[t][r] + bias) * wt;
    }

    // reduce across the 16 column-lanes within each half-wave
    #pragma unroll
    for (int r = 0; r < 8; ++r) {
        float v = rowsum[r];
        v += __shfl_xor(v, 1, 32);
        v += __shfl_xor(v, 2, 32);
        v += __shfl_xor(v, 4, 32);
        v += __shfl_xor(v, 8, 32);
        rowsum[r] = v;
    }

    if (m == 0) {
        #pragma unroll
        for (int r = 0; r < 8; ++r)
            atomicAdd(&ssum[kh * 8 + r], rowsum[r]);
    }
    __syncthreads();

    if (threadIdx.x < 16)
        scores[(size_t)b * FEATN + n0 + threadIdx.x] = ssum[threadIdx.x];
}

// ---------------------------------------------------------------------------
// Kernel 4: softmax over N=512 per batch row -> alpha
// ---------------------------------------------------------------------------
__global__ __launch_bounds__(512)
void softmax_kernel(const float* __restrict__ scores, float* __restrict__ alpha) {
    __shared__ float wred[16];
    __shared__ float bcast;
    const int b = blockIdx.x;
    const int n = threadIdx.x;
    const float s = scores[(size_t)b * FEATN + n];

    float m = s;
    m = fmaxf(m, __shfl_xor(m, 16, 32));
    m = fmaxf(m, __shfl_xor(m, 8, 32));
    m = fmaxf(m, __shfl_xor(m, 4, 32));
    m = fmaxf(m, __shfl_xor(m, 2, 32));
    m = fmaxf(m, __shfl_xor(m, 1, 32));
    if ((n & 31) == 0) wred[n >> 5] = m;
    __syncthreads();
    if (n == 0) {
        float g = wred[0];
        for (int i = 1; i < 16; ++i) g = fmaxf(g, wred[i]);
        bcast = g;
    }
    __syncthreads();
    const float gmax = bcast;

    const float e = fast_exp(s - gmax);
    float t = e;
    t += __shfl_xor(t, 16, 32);
    t += __shfl_xor(t, 8, 32);
    t += __shfl_xor(t, 4, 32);
    t += __shfl_xor(t, 2, 32);
    t += __shfl_xor(t, 1, 32);
    if ((n & 31) == 0) wred[n >> 5] = t;
    __syncthreads();
    if (n == 0) {
        float g = 0.0f;
        for (int i = 0; i < 16; ++i) g += wred[i];
        bcast = g;
    }
    __syncthreads();

    alpha[(size_t)b * FEATN + n] = e * __builtin_amdgcn_rcpf(bcast);
}

// ---------------------------------------------------------------------------
// Kernel 5: att_feats[b][f] = sum_n alpha[b][n] * feats[b][n][f]
// Grid: (FEATSZ/1024, BATCH), block 256, float4 per thread. Bandwidth bound.
// ---------------------------------------------------------------------------
__global__ __launch_bounds__(256)
void attsum_kernel(const float* __restrict__ feats,
                   const float* __restrict__ alpha,
                   float* __restrict__ att) {
    __shared__ float al[FEATN];
    const int b  = blockIdx.y;
    const int f0 = blockIdx.x * 1024 + threadIdx.x * 4;

    al[threadIdx.x]       = alpha[(size_t)b * FEATN + threadIdx.x];
    al[threadIdx.x + 256] = alpha[(size_t)b * FEATN + threadIdx.x + 256];
    __syncthreads();

    float4 acc = {0.0f, 0.0f, 0.0f, 0.0f};
    const float* base = feats + (size_t)b * FEATN * FEATSZ + f0;
    for (int n = 0; n < FEATN; ++n) {
        const float a = al[n];
        float4 v = *(const float4*)(base + (size_t)n * FEATSZ);
        acc.x = fmaf(a, v.x, acc.x);
        acc.y = fmaf(a, v.y, acc.y);
        acc.z = fmaf(a, v.z, acc.z);
        acc.w = fmaf(a, v.w, acc.w);
    }
    *(float4*)(att + (size_t)b * FEATSZ + f0) = acc;
}

// ---------------------------------------------------------------------------
// Launcher
// ---------------------------------------------------------------------------
extern "C" void kernel_launch(void* const* d_in, const int* in_sizes, int n_in,
                              void* d_out, int out_size, void* d_ws, size_t ws_size,
                              hipStream_t stream) {
    const float* feats = (const float*)d_in[0];   // (64, 512, 2048)
    const float* key   = (const float*)d_in[1];   // (64, 1024)
    const float* wh_w  = (const float*)d_in[2];   // (512, 1024)
    const float* wh_b  = (const float*)d_in[3];   // (512,)
    const float* wv_w  = (const float*)d_in[4];   // (512, 2048)
    const float* wv_b  = (const float*)d_in[5];   // (512,)
    const float* wa_w  = (const float*)d_in[6];   // (1, 512)

    float* att   = (float*)d_out;                       // (64, 2048)
    float* alpha = (float*)d_out + BATCH * FEATSZ;      // (64, 512)

    char* ws = (char*)d_ws;
    __bf16* wvw_bf = (__bf16*)ws;                                         // 2 MB
    float* hb     = (float*)(ws + (size_t)ATT * FEATSZ * 2);              // 128 KB
    float* scores = (float*)(ws + (size_t)ATT * FEATSZ * 2
                                + (size_t)BATCH * ATT * 4);               // 128 KB

    // 1) wv_w -> bf16 (262144 float4s)
    cvt_bf16_kernel<<<(ATT * FEATSZ / 4) / 256, 256, 0, stream>>>(wv_w, wvw_bf);
    // 2) hb = key @ wh_w^T + wh_b + wv_b
    hb_kernel<<<(BATCH * ATT) / 256, 256, 0, stream>>>(key, wh_w, wh_b, wv_b, hb);
    // 3) WMMA GEMM + tanh + wa dot -> scores
    scores_kernel<<<dim3(FEATN / 16, BATCH), 256, 0, stream>>>(feats, wvw_bf, hb, wa_w, scores);
    // 4) softmax -> alpha (written directly to d_out)
    softmax_kernel<<<BATCH, FEATN, 0, stream>>>(scores, alpha);
    // 5) alpha-weighted feats sum -> att_feats
    attsum_kernel<<<dim3(FEATSZ / 1024, BATCH), 256, 0, stream>>>(feats, alpha, att);
}
